// MultiHeadAttention_74715251081284
// MI455X (gfx1250) — matrix operand order repaired and checked by reading, not compiled, not run on hardware
//
#include <hip/hip_runtime.h>

// ---------------------------------------------------------------------------
// MHA forward for MI455X (gfx1250, wave32, WMMA f32_16x16x32_f16).
// Pipeline:
//   1) transpose+cvt weights f32 -> f16 N-major
//   2) QKV projections (WMMA GEMM, f32 A converted on the fly)
//   3) scores = Q K^T (scale folded into Q) + softmax  -> attn (f32, d_out)
//   4) attn @ V (WMMA, attn re-read from global/L2)    -> hOut (f32)
//   5) out = hOut @ Wo + bo (WMMA)                     -> d_out
// ---------------------------------------------------------------------------

typedef __attribute__((ext_vector_type(16))) _Float16 v16h;
typedef __attribute__((ext_vector_type(8)))  _Float16 v8h;
typedef __attribute__((ext_vector_type(8)))  float    v8f;

#define S_LEN   2048
#define D_MODEL 1024
#define NHEAD   16
#define HDIM    64
#define BATCH   4
#define BH      (BATCH * NHEAD)   // 64
#define MROWS   (BATCH * S_LEN)   // 8192

// ---- WMMA helper ----------------------------------------------------------
__device__ __forceinline__ v8f wmma16(v16h a, v16h b, v8f c) {
  return __builtin_amdgcn_wmma_f32_16x16x32_f16(
      /*neg_a=*/false, a, /*neg_b=*/false, b,
      /*c_mod=*/(short)0, c, /*reuse_a=*/false, /*reuse_b=*/false);
}

// A-fragment (16x32, f16) from row-major f32 source.
// Lane layout (ISA 7.12.2): lanes 0-15 row M=lane, elems 0..7 -> K 0..7,
// elems 8..15 -> K 16..23; lanes 16-31 same rows, K 8..15 / 24..31.
__device__ __forceinline__ v16h frag_a_f32(const float* __restrict__ A,
                                           long lda, int m0, int k0, int lane) {
  const int row = m0 + (lane & 15);
  const int kb  = (lane < 16) ? 0 : 8;
  const float* p = A + (long)row * lda + k0 + kb;
  v8f lo = *(const v8f*)(p);
  v8f hi = *(const v8f*)(p + 16);
  v16h f;
#pragma unroll
  for (int i = 0; i < 8; ++i) { f[i] = (_Float16)lo[i]; f[8 + i] = (_Float16)hi[i]; }
  return f;
}

// A-fragment from row-major f16 source.
__device__ __forceinline__ v16h frag_a_f16(const _Float16* __restrict__ A,
                                           long lda, int m0, int k0, int lane) {
  const int row = m0 + (lane & 15);
  const int kb  = (lane < 16) ? 0 : 8;
  const _Float16* p = A + (long)row * lda + k0;
  v8h lo = *(const v8h*)(p + kb);
  v8h hi = *(const v8h*)(p + kb + 16);
  v16h f;
#pragma unroll
  for (int i = 0; i < 8; ++i) { f[i] = lo[i]; f[8 + i] = hi[i]; }
  return f;
}

// B-fragment (32x16, f16) from N-major (transposed) f16 source: lane owns
// column n0+(lane&15); elems 0..7 -> K kb..kb+7, elems 8..15 -> K kb+16..kb+23
// (kb = lane<16 ? 0 : 8), matching the A-fragment layout.
__device__ __forceinline__ v16h frag_b_nmajor(const _Float16* __restrict__ Bt,
                                              long ldb, int n0, int k0, int lane) {
  const int n  = n0 + (lane & 15);
  const int kb = (lane < 16) ? 0 : 8;
  const _Float16* p = Bt + (long)n * ldb + k0 + kb;
  v8h lo = *(const v8h*)(p);
  v8h hi = *(const v8h*)(p + 16);
  v16h f;
#pragma unroll
  for (int i = 0; i < 8; ++i) { f[i] = lo[i]; f[8 + i] = hi[i]; }
  return f;
}

// ---- 1) weight transpose + f16 convert:  Wt[n*D+k] = W[k*D+n] -------------
__global__ void k_transpose_w_f16(const float* __restrict__ W,
                                  _Float16* __restrict__ Wt) {
  long i = (long)blockIdx.x * blockDim.x + threadIdx.x;   // over D*D
  if (i >= (long)D_MODEL * D_MODEL) return;
  int k = (int)(i >> 10), n = (int)(i & 1023);
  Wt[(long)n * D_MODEL + k] = (_Float16)W[i];
}

// ---- 2/5) generic WMMA GEMM: C[8192x1024] = A f32 @ Bt f16 + bias ---------
// mode 0/1: head-split f16 out (Q uses scale=1/32 folded in, K scale=1)
// mode 2  : head-split transposed f16 out (V -> Vt[bh][d][s])
// mode 3  : f32 row-major out (O projection into d_out)
__global__ __launch_bounds__(128)
void k_gemm(const float* __restrict__ A, const _Float16* __restrict__ Bt,
            const float* __restrict__ bias, _Float16* __restrict__ outH,
            float* __restrict__ outF, int mode, float scale) {
  const int lane = threadIdx.x & 31;
  const int wave = threadIdx.x >> 5;                 // 0..3
  const int m0   = blockIdx.x * 64 + wave * 16;      // gridDim.x = 128
  const int n0b  = blockIdx.y * 64;                  // gridDim.y = 16

  v8f acc[4];
  v8f z = {};
#pragma unroll
  for (int t = 0; t < 4; ++t) acc[t] = z;

  for (int k0 = 0; k0 < D_MODEL; k0 += 32) {
    if (k0 + 32 < D_MODEL)
      __builtin_prefetch(A + (long)(m0 + (lane & 15)) * D_MODEL + k0 + 32, 0, 0);
    v16h a = frag_a_f32(A, D_MODEL, m0, k0, lane);
#pragma unroll
    for (int t = 0; t < 4; ++t) {
      v16h b = frag_b_nmajor(Bt, D_MODEL, n0b + t * 16, k0, lane);
      acc[t] = wmma16(a, b, acc[t]);
    }
  }

  const int col   = lane & 15;
  const int rbase = (lane < 16) ? 0 : 8;
#pragma unroll
  for (int t = 0; t < 4; ++t) {
    const int n  = n0b + t * 16 + col;
    const float bv = bias[n];
#pragma unroll
    for (int r = 0; r < 8; ++r) {
      const int m = m0 + rbase + r;
      const float v = acc[t][r] + bv;
      if (mode == 3) {
        outF[(long)m * D_MODEL + n] = v;
      } else {
        const int b = m >> 11, s = m & (S_LEN - 1);
        const int h = n >> 6,  d = n & (HDIM - 1);
        const int bh = b * NHEAD + h;
        if (mode == 2)
          outH[((long)bh * HDIM + d) * S_LEN + s] = (_Float16)v;           // V^T
        else
          outH[((long)bh * S_LEN + s) * HDIM + d] = (_Float16)(v * scale); // Q/K
      }
    }
  }
}

// ---- 3) scores = Q K^T (+softmax), attn written f32 -----------------------
// Block = 8 waves; each wave owns 256 key columns (16 WMMA accumulators held
// in registers). Softmax: shfl_xor within half-waves + LDS across waves.
__global__ __launch_bounds__(256)
void k_scores_softmax(const _Float16* __restrict__ Qh,
                      const _Float16* __restrict__ Kh,
                      float* __restrict__ attn) {
  __shared__ float red[8][16];
  const int lane = threadIdx.x & 31;
  const int wave = threadIdx.x >> 5;         // 0..7
  const int qt   = blockIdx.x;               // 0..127
  const int bh   = blockIdx.y;               // 0..63
  const int m0   = qt * 16;
  const _Float16* Qb = Qh + (long)bh * S_LEN * HDIM;
  const _Float16* Kb = Kh + (long)bh * S_LEN * HDIM;

  v16h afr[4];
#pragma unroll
  for (int ds = 0; ds < 4; ++ds) afr[ds] = frag_a_f16(Qb, HDIM, m0, ds * 32, lane);

  v8f acc[16];
  v8f z = {};
#pragma unroll
  for (int t = 0; t < 16; ++t) acc[t] = z;

#pragma unroll
  for (int t = 0; t < 16; ++t) {
    const int n0 = wave * 256 + t * 16;
#pragma unroll
    for (int ds = 0; ds < 4; ++ds) {
      v16h b = frag_b_nmajor(Kb, HDIM, n0, ds * 32, lane);
      acc[t] = wmma16(afr[ds], b, acc[t]);
    }
  }

  const int rbase = (lane < 16) ? 0 : 8;

  // row max: per-lane over 16 tiles, shfl across the 16 lanes of a half-wave
  float rowmax[8];
#pragma unroll
  for (int r = 0; r < 8; ++r) {
    float m = -3.4e38f;
#pragma unroll
    for (int t = 0; t < 16; ++t) m = fmaxf(m, acc[t][r]);
#pragma unroll
    for (int off = 1; off < 16; off <<= 1) m = fmaxf(m, __shfl_xor(m, off, 32));
    rowmax[r] = m;
  }
  if ((lane & 15) == 0) {
#pragma unroll
    for (int r = 0; r < 8; ++r) red[wave][rbase + r] = rowmax[r];
  }
  __syncthreads();
#pragma unroll
  for (int r = 0; r < 8; ++r) {
    float m = red[0][rbase + r];
#pragma unroll
    for (int w = 1; w < 8; ++w) m = fmaxf(m, red[w][rbase + r]);
    rowmax[r] = m;
  }
  __syncthreads();

  // exp + row sum
  float rowsum[8];
#pragma unroll
  for (int r = 0; r < 8; ++r) {
    float s = 0.f;
#pragma unroll
    for (int t = 0; t < 16; ++t) {
      float e = __expf(acc[t][r] - rowmax[r]);
      acc[t][r] = e;
      s += e;
    }
#pragma unroll
    for (int off = 1; off < 16; off <<= 1) s += __shfl_xor(s, off, 32);
    rowsum[r] = s;
  }
  if ((lane & 15) == 0) {
#pragma unroll
    for (int r = 0; r < 8; ++r) red[wave][rbase + r] = rowsum[r];
  }
  __syncthreads();
  float rinv[8];
#pragma unroll
  for (int r = 0; r < 8; ++r) {
    float s = 0.f;
#pragma unroll
    for (int w = 0; w < 8; ++w) s += red[w][rbase + r];
    rinv[r] = 1.f / s;
  }

  // write normalized attn rows
  float* arow = attn + ((long)bh * S_LEN + m0) * S_LEN;
  const int col0 = wave * 256;
#pragma unroll
  for (int t = 0; t < 16; ++t) {
    const int c = col0 + t * 16 + (lane & 15);
#pragma unroll
    for (int r = 0; r < 8; ++r)
      arow[(long)(rbase + r) * S_LEN + c] = acc[t][r] * rinv[r];
  }
}

// ---- 4) hOut = attn @ V  (A: f32 attn from global, B: Vt f16 N-major) -----
__global__ __launch_bounds__(128)
void k_attnv(const float* __restrict__ attn, const _Float16* __restrict__ Vt,
             float* __restrict__ hOut) {
  const int lane = threadIdx.x & 31;
  const int wave = threadIdx.x >> 5;     // 0..3 -> 16-wide d-tile
  const int qt   = blockIdx.x;           // 0..127
  const int bh   = blockIdx.y;           // 0..63
  const int m0   = qt * 16;
  const float*    Ab = attn + (long)bh * S_LEN * S_LEN;
  const _Float16* Vb = Vt + (long)bh * HDIM * S_LEN;
  const int n0 = wave * 16;

  v8f acc = {};
  for (int k0 = 0; k0 < S_LEN; k0 += 32) {
    v16h a = frag_a_f32(Ab, S_LEN, m0, k0, lane);
    v16h b = frag_b_nmajor(Vb, S_LEN, n0, k0, lane);
    acc = wmma16(a, b, acc);
  }

  const int b = bh >> 4, h = bh & 15;
  const int d = n0 + (lane & 15);
  const int rbase = (lane < 16) ? 0 : 8;
#pragma unroll
  for (int r = 0; r < 8; ++r) {
    const int s = m0 + rbase + r;
    hOut[((long)b * S_LEN + s) * D_MODEL + h * HDIM + d] = acc[r];
  }
}

// ---------------------------------------------------------------------------
extern "C" void kernel_launch(void* const* d_in, const int* in_sizes, int n_in,
                              void* d_out, int out_size, void* d_ws, size_t ws_size,
                              hipStream_t stream) {
  const float* q  = (const float*)d_in[0];
  const float* k  = (const float*)d_in[1];
  const float* v  = (const float*)d_in[2];
  const float* Wq = (const float*)d_in[3];
  const float* bq = (const float*)d_in[4];
  const float* Wk = (const float*)d_in[5];
  const float* bk = (const float*)d_in[6];
  const float* Wv = (const float*)d_in[7];
  const float* bv = (const float*)d_in[8];
  const float* Wo = (const float*)d_in[9];
  const float* bo = (const float*)d_in[10];

  float* out  = (float*)d_out;
  float* attn = out + (long)BATCH * S_LEN * D_MODEL;   // (out, attn) tuple order

  // Workspace carve-up (~88 MB total)
  char* ws = (char*)d_ws;
  _Float16* WqT = (_Float16*)ws; ws += (long)D_MODEL * D_MODEL * 2;
  _Float16* WkT = (_Float16*)ws; ws += (long)D_MODEL * D_MODEL * 2;
  _Float16* WvT = (_Float16*)ws; ws += (long)D_MODEL * D_MODEL * 2;
  _Float16* WoT = (_Float16*)ws; ws += (long)D_MODEL * D_MODEL * 2;
  _Float16* Qh  = (_Float16*)ws; ws += (long)BH * S_LEN * HDIM * 2;
  _Float16* Kh  = (_Float16*)ws; ws += (long)BH * S_LEN * HDIM * 2;
  _Float16* Vt  = (_Float16*)ws; ws += (long)BH * S_LEN * HDIM * 2;
  float*    hOut = (float*)ws;   ws += (long)MROWS * D_MODEL * 4;

  // 1) weight transpose/convert
  {
    const int n = D_MODEL * D_MODEL;
    dim3 g((n + 255) / 256), b(256);
    k_transpose_w_f16<<<g, b, 0, stream>>>(Wq, WqT);
    k_transpose_w_f16<<<g, b, 0, stream>>>(Wk, WkT);
    k_transpose_w_f16<<<g, b, 0, stream>>>(Wv, WvT);
    k_transpose_w_f16<<<g, b, 0, stream>>>(Wo, WoT);
  }

  // 2) QKV projections (softmax scale 1/sqrt(D_MODEL)=1/32 folded into Q)
  dim3 gG(MROWS / 64, D_MODEL / 64), bG(128);
  k_gemm<<<gG, bG, 0, stream>>>(q, WqT, bq, Qh, nullptr, 0, 0.03125f);
  k_gemm<<<gG, bG, 0, stream>>>(k, WkT, bk, Kh, nullptr, 1, 1.0f);
  k_gemm<<<gG, bG, 0, stream>>>(v, WvT, bv, Vt, nullptr, 2, 1.0f);

  // 3) scores + softmax -> attn (f32 in d_out)
  k_scores_softmax<<<dim3(S_LEN / 16, BH), 256, 0, stream>>>(Qh, Kh, attn);

  // 4) attn @ V -> merged head output (f32)
  k_attnv<<<dim3(S_LEN / 16, BH), 128, 0, stream>>>(attn, Vt, hOut);

  // 5) output projection -> d_out
  k_gemm<<<gG, bG, 0, stream>>>(hOut, WoT, bo, nullptr, out, 3, 1.0f);
}
